// MultiHeadSeqAttention_66941360275804
// MI455X (gfx1250) — matrix operand order repaired and checked by reading, not compiled, not run on hardware
//
#include <hip/hip_runtime.h>
#include <hip/hip_bf16.h>

// ---------------------------------------------------------------------------
// Problem constants (match reference)
// ---------------------------------------------------------------------------
#define BB    4
#define MM    2048
#define HH    512
#define KHEAD 8
#define DD    64
#define SPAN  512
#define TK    (MM + SPAN)   // 2560 key/value rows per batch

typedef __attribute__((ext_vector_type(16))) __bf16 v16bf;
typedef __attribute__((ext_vector_type(8)))  __bf16 v8bf;
typedef __attribute__((ext_vector_type(8)))  float  v8f;
typedef __attribute__((ext_vector_type(4)))  float  v4f;

union V16U { v16bf v; v8bf h[2]; };

// Async global -> LDS copy of 32 bytes (two B128 ops), CDNA5 async path.
__device__ __forceinline__ void async_copy32(unsigned lds_addr,
                                             const __bf16* gptr) {
  asm volatile(
      "global_load_async_to_lds_b128 %0, %1, off\n\t"
      "global_load_async_to_lds_b128 %0, %1, off offset:16"
      :
      : "v"(lds_addr), "v"((unsigned long long)gptr)
      : "memory");
}
__device__ __forceinline__ void async_wait0() {
  asm volatile("s_wait_asynccnt 0" ::: "memory");
}

// ---------------------------------------------------------------------------
// fp32 -> bf16 conversion (grid-stride)
// ---------------------------------------------------------------------------
__global__ void cvt_f32_bf16(const float* __restrict__ src,
                             __bf16* __restrict__ dst, int n) {
  for (int i = blockIdx.x * blockDim.x + threadIdx.x; i < n;
       i += gridDim.x * blockDim.x)
    dst[i] = (__bf16)src[i];
}

// key_pe [D, SPAN] fp32 -> transposed bf16 [SPAN, D] so attention B-fragments
// (16 contiguous K=d values per lane) are contiguous 32B loads.
__global__ void cvt_pe_transpose(const float* __restrict__ src,
                                 __bf16* __restrict__ dst) {
  int idx = blockIdx.x * blockDim.x + threadIdx.x;   // over D*SPAN
  if (idx < DD * SPAN) {
    int d = idx >> 9;          // /SPAN
    int s = idx & (SPAN - 1);
    dst[s * DD + d] = (__bf16)src[idx];
  }
}

// ---------------------------------------------------------------------------
// Tiled WMMA GEMM:  Y[N,Hout] = X[N,Hin] @ W[Hout,Hin]^T   (bf16 in, f32 acc)
// Block = 256 threads (8 waves). Block tile = 128x128. Wave slab = 32x64
// (2 m-tiles x 4 n-tiles -> 8 WMMAs per K=32 step). Double-buffered LDS
// staged with global_load_async_to_lds_b128: copies for step kk+1 overlap
// the WMMAs of step kk; one barrier per K-step.
// ---------------------------------------------------------------------------
template <typename OutT>
__global__ __launch_bounds__(256) void gemm_bf16_wmma(
    const __bf16* __restrict__ X, const __bf16* __restrict__ W,
    OutT* __restrict__ Y, int N, int Hin, int Hout) {
  __shared__ __bf16 Xs[2][128][32];
  __shared__ __bf16 Ws[2][128][32];

  const int tid  = threadIdx.x;
  const int lane = tid & 31;
  const int wave = tid >> 5;          // 8 waves
  const int wmi  = wave >> 1;         // 0..3 -> 32-row slab
  const int wni  = wave & 1;          // 0..1 -> 64-col slab
  const int gm0  = blockIdx.y * 128;  // output rows
  const int gn0  = blockIdx.x * 128;  // output cols

  v8f acc[2][4];
#pragma unroll
  for (int mt = 0; mt < 2; ++mt)
#pragma unroll
    for (int nt = 0; nt < 4; ++nt)
      acc[mt][nt] = (v8f){0.f, 0.f, 0.f, 0.f, 0.f, 0.f, 0.f, 0.f};

  // per-thread staging chunk: 16 elems (32B) of each tile
  const int e  = tid << 4;            // 256 threads * 16 = 4096 elems
  const int sr = e >> 5, scn = e & 31;
  const unsigned ldsX[2] = {
      (unsigned)(unsigned long long)&Xs[0][sr][scn],
      (unsigned)(unsigned long long)&Xs[1][sr][scn]};
  const unsigned ldsW[2] = {
      (unsigned)(unsigned long long)&Ws[0][sr][scn],
      (unsigned)(unsigned long long)&Ws[1][sr][scn]};
  const __bf16* gx = &X[(size_t)(gm0 + sr) * Hin + scn];
  const __bf16* gw = &W[(size_t)(gn0 + sr) * Hin + scn];

  const int nk = Hin >> 5;            // K steps of 32

  // prologue: stage tile 0 into buffer 0
  async_copy32(ldsX[0], gx);
  async_copy32(ldsW[0], gw);

  for (int kk = 0; kk < nk; ++kk) {
    const int cur = kk & 1;
    async_wait0();                    // buffer `cur` landed (this wave)
    __syncthreads();                  // ... for all waves; prior reads retired

    if (kk + 1 < nk) {                // overlap next tile's copy with WMMAs
      const int k1 = (kk + 1) << 5;
      async_copy32(ldsX[cur ^ 1], gx + k1);
      async_copy32(ldsW[cur ^ 1], gw + k1);
    }

    V16U a[2];
#pragma unroll
    for (int mt = 0; mt < 2; ++mt) {
      int r  = (wmi << 5) + (mt << 4) + (lane & 15);
      int kb = (lane >> 4) << 4;
      a[mt].h[0] = *(const v8bf*)&Xs[cur][r][kb];
      a[mt].h[1] = *(const v8bf*)&Xs[cur][r][kb + 8];
    }
#pragma unroll
    for (int nt = 0; nt < 4; ++nt) {
      int c  = (wni << 6) + (nt << 4) + (lane & 15);
      int ks = (lane >> 4) << 4;
      v16bf b = *(const v16bf*)&Ws[cur][c][ks];
#pragma unroll
      for (int mt = 0; mt < 2; ++mt)
        acc[mt][nt] = __builtin_amdgcn_wmma_f32_16x16x32_bf16(
            false, a[mt].v, false, b, (short)0, acc[mt][nt], false, false);
    }
  }

  // C/D layout: lanes 0-15 N=lane rows M=vg; lanes 16-31 N=lane-16 rows M=8+vg
  const int cb = gn0 + (wni << 6) + (lane & 15);
#pragma unroll
  for (int mt = 0; mt < 2; ++mt) {
    const int rbase = gm0 + (wmi << 5) + (mt << 4) + ((lane >> 4) << 3);
#pragma unroll
    for (int nt = 0; nt < 4; ++nt)
#pragma unroll
      for (int vg = 0; vg < 8; ++vg)
        Y[(size_t)(rbase + vg) * Hout + cb + (nt << 4)] =
            (OutT)acc[mt][nt][vg];
  }
}

// ---------------------------------------------------------------------------
// Banded attention for one (batch, head, 16-row tile) per wave (block=32).
//   scores cont: S[i][l] = q[m0+i] . k[m0+l], l = 0..527  (33 WMMA col-tiles)
//   band+pos   : S[i][s] = S[i][i+s] + q[i].pe[:,s]   (pos WMMA with band as C)
//   softmax (2 lanes/row, float4, shfl_xor reduce), skewed bf16 prob matrix,
//   then AV via WMMA over K=544.
// ---------------------------------------------------------------------------
__global__ __launch_bounds__(32) void banded_attn_wmma(
    const __bf16* __restrict__ Q,   // [B*M, H]   (heads interleaved: h=kh*64+d)
    const __bf16* __restrict__ Km,  // [B*TK, H]
    const __bf16* __restrict__ Vm,  // [B*TK, H]
    const __bf16* __restrict__ PeT, // [SPAN, D]
    __bf16* __restrict__ Out) {     // [B*M, H]
  __shared__ float  Sc[16][544];    // cont scores / band / probabilities
  __shared__ __bf16 At[16][544];    // skewed bf16 probs (K dim for AV)
  __shared__ __bf16 Vt[64][32];     // V tile transposed [d][k]

  const int lane  = threadIdx.x;
  const int tiles = MM / 16;
  const int head  = blockIdx.x / tiles;
  const int mt    = blockIdx.x % tiles;
  const int b     = head / KHEAD;
  const int kh    = head % KHEAD;
  const int m0    = mt * 16;

  const size_t qrow0 = (size_t)(b * MM + m0) * HH + kh * DD;
  const size_t krow0 = (size_t)(b * TK + m0) * HH + kh * DD;

  // --- Q A-fragments (16 x 64 -> two K=32 fragments), kept in registers ----
  V16U aq[2];
  {
    const size_t r = qrow0 + (size_t)(lane & 15) * HH;
#pragma unroll
    for (int c = 0; c < 2; ++c) {
      int kb     = c * 32 + ((lane >> 4) << 4);
      aq[c].h[0] = *(const v8bf*)&Q[r + kb];
      aq[c].h[1] = *(const v8bf*)&Q[r + kb + 8];
    }
  }

  // --- content scores: 33 tiles of 16 key columns -------------------------
  for (int kt = 0; kt < 33; ++kt) {
    v8f c = (v8f){0.f, 0.f, 0.f, 0.f, 0.f, 0.f, 0.f, 0.f};
    const size_t kr = krow0 + (size_t)(kt * 16 + (lane & 15)) * HH;
#pragma unroll
    for (int cc = 0; cc < 2; ++cc) {
      v16bf bfr = *(const v16bf*)&Km[kr + cc * 32 + ((lane >> 4) << 4)];
      c = __builtin_amdgcn_wmma_f32_16x16x32_bf16(false, aq[cc].v, false, bfr,
                                                  (short)0, c, false, false);
    }
    const int ro = (lane >> 4) << 3, co = kt * 16 + (lane & 15);
#pragma unroll
    for (int vg = 0; vg < 8; ++vg) Sc[ro + vg][co] = c[vg];
  }
  __syncthreads();

  // --- band extraction fused with positional term (band as WMMA C) --------
  for (int st = 0; st < 32; ++st) {
    v8f c;
    const int ro = (lane >> 4) << 3, nn = lane & 15;
#pragma unroll
    for (int vg = 0; vg < 8; ++vg) {
      int row = ro + vg;
      c[vg]   = Sc[row][row + st * 16 + nn];   // skew gather
    }
    const size_t pr = (size_t)(st * 16 + nn) * DD;
#pragma unroll
    for (int cc = 0; cc < 2; ++cc) {
      v16bf bfr = *(const v16bf*)&PeT[pr + cc * 32 + ((lane >> 4) << 4)];
      c = __builtin_amdgcn_wmma_f32_16x16x32_bf16(false, aq[cc].v, false, bfr,
                                                  (short)0, c, false, false);
    }
#pragma unroll
    for (int vg = 0; vg < 8; ++vg) Sc[ro + vg][st * 16 + nn] = c[vg];
  }
  __syncthreads();

  // --- zero skewed prob matrix --------------------------------------------
  for (int i = lane; i < 16 * 544; i += 32) (&At[0][0])[i] = (__bf16)0.f;
  __syncthreads();

  // --- softmax: 2 lanes per row, 256 cols each, float4 vectorized ---------
  {
    const float scale = 0.125f;            // 1/sqrt(64)
    const int   row   = lane & 15;
    const int   s0    = (lane >> 4) << 8;  // 0 or 256
    float mx = -3.4e38f;
    for (int s = 0; s < 256; s += 4) {
      v4f v = *(const v4f*)&Sc[row][s0 + s];
      mx = fmaxf(mx, fmaxf(fmaxf(v[0], v[1]), fmaxf(v[2], v[3])));
    }
    mx = fmaxf(mx, __shfl_xor(mx, 16));
    const float mxs = mx * scale;
    float sum = 0.f;
    for (int s = 0; s < 256; s += 4) {
      v4f v = *(const v4f*)&Sc[row][s0 + s];
      v4f e;
#pragma unroll
      for (int j = 0; j < 4; ++j) {
        e[j] = __expf(v[j] * scale - mxs);
        sum += e[j];
      }
      *(v4f*)&Sc[row][s0 + s] = e;
    }
    sum += __shfl_xor(sum, 16);
    const float inv = 1.f / sum;
    for (int s = 0; s < 256; ++s)
      At[row][row + s0 + s] = (__bf16)(Sc[row][s0 + s] * inv);   // re-skew
  }
  __syncthreads();

  // --- AV: out[16,64] = At[16,544] @ V[544,64]  (17 K-steps of 32) --------
  v8f acc[4];
#pragma unroll
  for (int t = 0; t < 4; ++t)
    acc[t] = (v8f){0.f, 0.f, 0.f, 0.f, 0.f, 0.f, 0.f, 0.f};

  for (int kc = 0; kc < 17; ++kc) {
    // stage V rows [m0+kc*32, +32) x 64 transposed into Vt[d][k]
    for (int i = lane; i < 32 * 64; i += 32) {
      int j = i >> 6, d = i & 63;
      int t = m0 + kc * 32 + j;
      Vt[d][j] = (t < TK) ? Vm[(size_t)(b * TK + t) * HH + kh * DD + d]
                          : (__bf16)0.f;
    }
    __syncthreads();

    V16U a;
    {
      int r  = lane & 15;
      int kb = kc * 32 + ((lane >> 4) << 4);
      a.h[0] = *(const v8bf*)&At[r][kb];
      a.h[1] = *(const v8bf*)&At[r][kb + 8];
    }
#pragma unroll
    for (int nt = 0; nt < 4; ++nt) {
      int dcol = (nt << 4) + (lane & 15);
      v16bf bfr = *(const v16bf*)&Vt[dcol][(lane >> 4) << 4];
      acc[nt] = __builtin_amdgcn_wmma_f32_16x16x32_bf16(
          false, a.v, false, bfr, (short)0, acc[nt], false, false);
    }
    __syncthreads();
  }

  // --- store head output (bf16) -------------------------------------------
  const int ro = (lane >> 4) << 3, co = lane & 15;
#pragma unroll
  for (int nt = 0; nt < 4; ++nt) {
#pragma unroll
    for (int vg = 0; vg < 8; ++vg)
      Out[(size_t)(b * MM + m0 + ro + vg) * HH + kh * DD + (nt << 4) + co] =
          (__bf16)acc[nt][vg];
  }
}

// ---------------------------------------------------------------------------
// Host-side launch
// ---------------------------------------------------------------------------
extern "C" void kernel_launch(void* const* d_in, const int* in_sizes, int n_in,
                              void* d_out, int out_size, void* d_ws,
                              size_t ws_size, hipStream_t stream) {
  const float* query  = (const float*)d_in[0];  // [B, M, H]
  const float* key    = (const float*)d_in[1];  // [B, M+SPAN, H]
  const float* value  = (const float*)d_in[2];  // [B, M+SPAN, H]
  const float* key_pe = (const float*)d_in[3];  // [1, D, SPAN]
  const float* Wq     = (const float*)d_in[4];  // [H, H]
  const float* Wk     = (const float*)d_in[5];
  const float* Wv     = (const float*)d_in[6];
  const float* Wo     = (const float*)d_in[7];
  float* out = (float*)d_out;                   // [B, M, H] fp32

  const int NQ  = BB * MM;   // 8192
  const int NKV = BB * TK;   // 10240

  size_t off = 0;
  auto alloc = [&](size_t bytes) {
    void* p = (char*)d_ws + off;
    off += (bytes + 255) & ~(size_t)255;
    return p;
  };
  __bf16* qbf = (__bf16*)alloc((size_t)NQ * HH * 2);
  __bf16* kbf = (__bf16*)alloc((size_t)NKV * HH * 2);
  __bf16* vbf = (__bf16*)alloc((size_t)NKV * HH * 2);
  __bf16* wqb = (__bf16*)alloc((size_t)HH * HH * 2);
  __bf16* wkb = (__bf16*)alloc((size_t)HH * HH * 2);
  __bf16* wvb = (__bf16*)alloc((size_t)HH * HH * 2);
  __bf16* wob = (__bf16*)alloc((size_t)HH * HH * 2);
  __bf16* peb = (__bf16*)alloc((size_t)SPAN * DD * 2);
  __bf16* qp  = (__bf16*)alloc((size_t)NQ * HH * 2);
  __bf16* kp  = (__bf16*)alloc((size_t)NKV * HH * 2);
  __bf16* vp  = (__bf16*)alloc((size_t)NKV * HH * 2);
  __bf16* ao  = (__bf16*)alloc((size_t)NQ * HH * 2);
  (void)ws_size; (void)in_sizes; (void)n_in; (void)out_size;

  // --- convert to bf16 -----------------------------------------------------
  auto cvt = [&](const float* s, __bf16* d, int n) {
    cvt_f32_bf16<<<(n + 255) / 256, 256, 0, stream>>>(s, d, n);
  };
  cvt(query, qbf, NQ * HH);
  cvt(key,   kbf, NKV * HH);
  cvt(value, vbf, NKV * HH);
  cvt(Wq, wqb, HH * HH);
  cvt(Wk, wkb, HH * HH);
  cvt(Wv, wvb, HH * HH);
  cvt(Wo, wob, HH * HH);
  cvt_pe_transpose<<<(DD * SPAN + 255) / 256, 256, 0, stream>>>(key_pe, peb);

  // --- projections: X @ W^T ------------------------------------------------
  gemm_bf16_wmma<__bf16><<<dim3(HH / 128, NQ / 128), 256, 0, stream>>>(
      qbf, wqb, qp, NQ, HH, HH);
  gemm_bf16_wmma<__bf16><<<dim3(HH / 128, NKV / 128), 256, 0, stream>>>(
      kbf, wkb, kp, NKV, HH, HH);
  gemm_bf16_wmma<__bf16><<<dim3(HH / 128, NKV / 128), 256, 0, stream>>>(
      vbf, wvb, vp, NKV, HH, HH);

  // --- banded attention: one wave per (head, 16-row tile) ------------------
  banded_attn_wmma<<<(BB * KHEAD) * (MM / 16), 32, 0, stream>>>(
      qp, kp, vp, peb, ao);

  // --- output projection: ao @ Wo^T -> fp32 --------------------------------
  gemm_bf16_wmma<float><<<dim3(HH / 128, NQ / 128), 256, 0, stream>>>(
      ao, wob, out, NQ, HH, HH);
}
